// MultiBoxLoss_89824946028596
// MI455X (gfx1250) — compile-verified, matrix operand order checked by reading, not run on hardware
//
#include <hip/hip_runtime.h>
#include <hip/hip_bf16.h>

// Problem constants (match reference)
#define BB 128
#define PP 8732
#define MM 16
#define CC 21
#define THRESH 0.5f
#define NEGPOS 3
#define TPB 256
#define TILES_PER_B ((PP + 15) / 16)   // 546
#define NTILES (BB * TILES_PER_B)      // 69888
#define WAVES_PER_BLK 8                // 256 / 32

typedef __attribute__((ext_vector_type(16))) _Float16 v16h;
typedef __attribute__((ext_vector_type(8)))  float    v8f;

// ---------------------------------------------------------------------------
// Kernel 1: per-image matching. One block (256 threads) per image.
//   ws_ovl[b*P+p] : overlap_each_prior (float)
//   ws_obj[b*P+p] : object_each_prior -> overwritten with gt_label (int)
//   ws_loc[b*P+p] : encoded gt loc (float4)
// ---------------------------------------------------------------------------
__global__ void __launch_bounds__(TPB)
ssd_match_kernel(const float* __restrict__ boxes,     // [B,M,4] xy
                 const int*   __restrict__ labels,    // [B,M]
                 const float* __restrict__ priors,    // [P,4] cxcy
                 float* __restrict__ ws_ovl,
                 int*   __restrict__ ws_obj,
                 float4* __restrict__ ws_loc)
{
    __shared__ float s_box[MM * 4];
    __shared__ int   s_lab[MM];
    __shared__ float s_oval[MM * TPB];
    __shared__ int   s_oidx[MM * TPB];
    __shared__ int   s_peo[MM];

    const int b   = blockIdx.x;
    const int tid = threadIdx.x;
    const size_t bp = (size_t)b * PP;

    if (tid < MM * 4) s_box[tid] = boxes[(size_t)b * MM * 4 + tid];
    if (tid < MM)     s_lab[tid] = labels[(size_t)b * MM + tid];
    __syncthreads();

    // per-thread per-object best prior (val, idx)
    float bov[MM];
    int   bop[MM];
#pragma unroll
    for (int m = 0; m < MM; ++m) { bov[m] = -1.0f; bop[m] = PP; }

    // Pass 1: per-prior max/argmax over objects + per-object candidates
    for (int p = tid; p < PP; p += TPB) {
        const float4 pr = ((const float4*)priors)[p];
        const float px0 = pr.x - pr.z * 0.5f, py0 = pr.y - pr.w * 0.5f;
        const float px1 = pr.x + pr.z * 0.5f, py1 = pr.y + pr.w * 0.5f;
        const float areaB = pr.z * pr.w;
        float best = -1.0f; int besti = 0;
#pragma unroll
        for (int m = 0; m < MM; ++m) {
            const float bx0 = s_box[4*m+0], by0 = s_box[4*m+1];
            const float bx1 = s_box[4*m+2], by1 = s_box[4*m+3];
            const float ix = fminf(bx1, px1) - fmaxf(bx0, px0);
            const float iy = fminf(by1, py1) - fmaxf(by0, py0);
            const float inter = fmaxf(ix, 0.f) * fmaxf(iy, 0.f);
            const float areaA = (bx1 - bx0) * (by1 - by0);
            const float ov = inter / (areaA + areaB - inter);
            if (ov > best) { best = ov; besti = m; }          // first-max tiebreak
            if (ov > bov[m]) { bov[m] = ov; bop[m] = p; }     // smallest p on ties
        }
        ws_ovl[bp + p] = best;
        ws_obj[bp + p] = besti;
    }
#pragma unroll
    for (int m = 0; m < MM; ++m) {
        s_oval[m * TPB + tid] = bov[m];
        s_oidx[m * TPB + tid] = bop[m];
    }
    __syncthreads();

    // reduce per-object best prior (threads 0..15)
    if (tid < MM) {
        float bv = -1.0f; int bi = PP;
        for (int t = 0; t < TPB; ++t) {
            const float v = s_oval[tid * TPB + t];
            const int   i = s_oidx[tid * TPB + t];
            if (v > bv || (v == bv && i < bi)) { bv = v; bi = i; }
        }
        s_peo[tid] = bi;
    }
    __syncthreads();

    // force each object to own its best prior (sequential: last write wins)
    if (tid == 0) {
        for (int m = 0; m < MM; ++m) {
            const int p = s_peo[m];
            ws_obj[bp + p] = m;
            ws_ovl[bp + p] = 1.0f;
        }
    }
    __syncthreads();

    // Pass 2: labels (threshold) + gcxgcy encode; overwrite ws_obj with label
    for (int p = tid; p < PP; p += TPB) {
        const float ov  = ws_ovl[bp + p];
        const int   obj = ws_obj[bp + p];
        const int   lab = (ov < THRESH) ? 0 : s_lab[obj];
        ws_obj[bp + p] = lab;

        const float bx0 = s_box[4*obj+0], by0 = s_box[4*obj+1];
        const float bx1 = s_box[4*obj+2], by1 = s_box[4*obj+3];
        const float cx = (bx0 + bx1) * 0.5f, cy = (by0 + by1) * 0.5f;
        const float w  = bx1 - bx0,          h  = by1 - by0;
        const float4 pr = ((const float4*)priors)[p];
        float4 g;
        g.x = (cx - pr.x) / (pr.z * 0.1f);
        g.y = (cy - pr.y) / (pr.w * 0.1f);
        g.z = __logf(w / pr.z) * 5.0f;
        g.w = __logf(h / pr.w) * 5.0f;
        ws_loc[bp + p] = g;
    }
}

// ---------------------------------------------------------------------------
// Kernel 2: per-prior cross entropy via WMMA row-sum of exp(scores).
// One wave handles a 16-prior tile: D = exp(S[16x32]) x ones[32x16] (f16 WMMA).
// conf_all[b*P+p] = log(sum_exp) - score[label]   (written into ws_ovl buffer)
//
// A-matrix 16x32 f16 layout (ISA 7.12.2): lanes 0-15 hold K 0..7 / 16..23,
// lanes 16-31 hold K 8..15 / 24..31 for row M = lane&15.
// Out-of-range priors (last tile only) are clamped for loads: a padded ROW
// only corrupts its own (unused) row sum, never a valid one. Class padding
// (c >= 21) must be exactly zero since it shares a row with valid classes.
// ---------------------------------------------------------------------------
__global__ void __launch_bounds__(TPB)
ssd_conf_kernel(const float* __restrict__ scores,   // [B,P,C]
                const int*   __restrict__ gtlab,    // [B,P] (ws_obj)
                float* __restrict__ conf_all)       // [B,P] (ws_ovl reused)
{
    __shared__ float s_se[WAVES_PER_BLK][16];

    const int lane = threadIdx.x & 31;
    const int wave = threadIdx.x >> 5;
    const int w    = blockIdx.x * WAVES_PER_BLK + wave;   // exact grid: w < NTILES

    const int b  = w / TILES_PER_B;
    const int p0 = (w % TILES_PER_B) * 16;
    const int m  = lane & 15;                 // row owned by this lane (A matrix)
    int p = p0 + m;
    if (p > PP - 1) p = PP - 1;               // clamp loads; padded rows unused
    const size_t srow = ((size_t)b * PP + p) * CC;

    // First 8 K-slots: classes 0..7 (lanes 0-15) or 8..15 (lanes 16-31).
    // Unconditional loads -> batched vmem, single wait.
    const int off0 = (lane < 16) ? 0 : 8;
    v16h a;
#pragma unroll
    for (int j = 0; j < 8; ++j)
        a[j] = (_Float16)__expf(scores[srow + off0 + j]);
    // Second 8 K-slots: classes 16..23 (lanes 0-15, valid 16..20) or 24..31
    // (lanes 16-31, all padding).
#pragma unroll
    for (int j = 0; j < 8; ++j) a[8 + j] = (_Float16)0.0f;
    if (lane < 16) {
#pragma unroll
        for (int j = 0; j < 5; ++j)
            a[8 + j] = (_Float16)__expf(scores[srow + 16 + j]);
    }

    v16h ones;
#pragma unroll
    for (int j = 0; j < 16; ++j) ones[j] = (_Float16)1.0f;

    v8f c = {};
    // D[m][n] = sum_k exp(S[m][k])  (replicated across all 16 columns)
    c = __builtin_amdgcn_wmma_f32_16x16x32_f16(
        /*neg_a=*/false, a, /*neg_b=*/false, ones,
        /*c_mod=*/(short)0, c, /*reuse_a=*/false, /*reuse_b=*/false);

    // D layout: lanes 0-15 hold rows 0-7 (vgpr r = row r), lanes 16-31 hold
    // rows 8-15. Columns identical -> lane 0 / lane 16 stage 16 row sums.
    if ((lane & 15) == 0) {
        const int r0 = (lane == 0) ? 0 : 8;
#pragma unroll
        for (int j = 0; j < 8; ++j) s_se[wave][r0 + j] = c[j];
    }
    __syncthreads();   // uniform: no early returns anywhere in this kernel

    if (lane < 16) {
        const int pown = p0 + lane;
        if (pown < PP) {
            const size_t idx = (size_t)b * PP + pown;
            const int   lab  = gtlab[idx];
            const float slab = scores[idx * CC + lab];
            conf_all[idx] = __logf(s_se[wave][lane]) - slab;
        }
    }
}

// ---------------------------------------------------------------------------
// Kernel 3: per-image loss. One block (256 threads) per image.
// conf_pos + top-k(3*n_pos) negatives via threshold bisection + smooth-L1.
// ---------------------------------------------------------------------------
__global__ void __launch_bounds__(TPB)
ssd_loss_kernel(const float*  __restrict__ pred_locs, // [B,P,4]
                const float*  __restrict__ conf_all,  // [B,P]
                const int*    __restrict__ gtlab,     // [B,P]
                const float4* __restrict__ gt_locs,   // [B,P]
                float* __restrict__ batch_total)      // [B]
{
    __shared__ float s_neg[PP];
    __shared__ float s_rf[TPB];
    __shared__ int   s_ri[TPB];

    const int b   = blockIdx.x;
    const int tid = threadIdx.x;
    const size_t bp = (size_t)b * PP;

    float conf_pos = 0.f, loc = 0.f;
    int   npos = 0;

    for (int p = tid; p < PP; p += TPB) {
        if (p + TPB < PP) {
            __builtin_prefetch(&conf_all[bp + p + TPB], 0, 1);  // global_prefetch_b8
        }
        const float cv = conf_all[bp + p];
        const int   lb = gtlab[bp + p];
        if (lb != 0) {
            ++npos;
            conf_pos += cv;
            const float4 pl = ((const float4*)pred_locs)[bp + p];
            const float4 gl = gt_locs[bp + p];
            const float d0 = fabsf(pl.x - gl.x), d1 = fabsf(pl.y - gl.y);
            const float d2 = fabsf(pl.z - gl.z), d3 = fabsf(pl.w - gl.w);
            loc += (d0 < 1.f ? 0.5f * d0 * d0 : d0 - 0.5f);
            loc += (d1 < 1.f ? 0.5f * d1 * d1 : d1 - 0.5f);
            loc += (d2 < 1.f ? 0.5f * d2 * d2 : d2 - 0.5f);
            loc += (d3 < 1.f ? 0.5f * d3 * d3 : d3 - 0.5f);
            s_neg[p] = 0.f;
        } else {
            s_neg[p] = cv;
        }
    }

    // deterministic tree reductions
    s_rf[tid] = conf_pos; __syncthreads();
    for (int o = TPB / 2; o > 0; o >>= 1) { if (tid < o) s_rf[tid] += s_rf[tid + o]; __syncthreads(); }
    conf_pos = s_rf[0]; __syncthreads();

    s_rf[tid] = loc; __syncthreads();
    for (int o = TPB / 2; o > 0; o >>= 1) { if (tid < o) s_rf[tid] += s_rf[tid + o]; __syncthreads(); }
    loc = s_rf[0]; __syncthreads();

    s_ri[tid] = npos; __syncthreads();
    for (int o = TPB / 2; o > 0; o >>= 1) { if (tid < o) s_ri[tid] += s_ri[tid + o]; __syncthreads(); }
    npos = s_ri[0]; __syncthreads();

    int kk = NEGPOS * npos;
    if (kk > PP) kk = PP;

    // block max of negatives
    float mx = 0.f;
    for (int p = tid; p < PP; p += TPB) mx = fmaxf(mx, s_neg[p]);
    s_rf[tid] = mx; __syncthreads();
    for (int o = TPB / 2; o > 0; o >>= 1) { if (tid < o) s_rf[tid] = fmaxf(s_rf[tid], s_rf[tid + o]); __syncthreads(); }
    mx = s_rf[0]; __syncthreads();

    // bisection for the kk-th largest threshold
    float lo = 0.f, hi = mx;
    for (int it = 0; it < 40; ++it) {
        const float mid = 0.5f * (lo + hi);
        int cnt = 0;
        for (int p = tid; p < PP; p += TPB) cnt += (s_neg[p] > mid) ? 1 : 0;
        s_ri[tid] = cnt; __syncthreads();
        for (int o = TPB / 2; o > 0; o >>= 1) { if (tid < o) s_ri[tid] += s_ri[tid + o]; __syncthreads(); }
        cnt = s_ri[0]; __syncthreads();
        if (cnt > kk) lo = mid; else hi = mid;
    }
    const float t = hi;

    // sum(v > t) and count(v > t); tie-correct with (kk - cnt) * t
    float S = 0.f; int cnt = 0;
    for (int p = tid; p < PP; p += TPB) {
        const float v = s_neg[p];
        if (v > t) { S += v; ++cnt; }
    }
    s_rf[tid] = S; s_ri[tid] = cnt; __syncthreads();
    for (int o = TPB / 2; o > 0; o >>= 1) {
        if (tid < o) { s_rf[tid] += s_rf[tid + o]; s_ri[tid] += s_ri[tid + o]; }
        __syncthreads();
    }
    S = s_rf[0]; cnt = s_ri[0];

    if (tid == 0) {
        const float conf_neg = S + (float)(kk - cnt) * t;
        batch_total[b] = (conf_pos + conf_neg + loc) / (float)npos;
    }
}

// ---------------------------------------------------------------------------
// Kernel 4: mean over batch.
// ---------------------------------------------------------------------------
__global__ void __launch_bounds__(BB)
ssd_mean_kernel(const float* __restrict__ batch_total, float* __restrict__ out)
{
    __shared__ float s[BB];
    const int tid = threadIdx.x;
    s[tid] = batch_total[tid];
    __syncthreads();
    for (int o = BB / 2; o > 0; o >>= 1) { if (tid < o) s[tid] += s[tid + o]; __syncthreads(); }
    if (tid == 0) out[0] = s[0] / (float)BB;
}

// ---------------------------------------------------------------------------
extern "C" void kernel_launch(void* const* d_in, const int* in_sizes, int n_in,
                              void* d_out, int out_size, void* d_ws, size_t ws_size,
                              hipStream_t stream) {
    const float* pred_locs   = (const float*)d_in[0];
    const float* pred_scores = (const float*)d_in[1];
    const float* boxes       = (const float*)d_in[2];
    const int*   labels      = (const int*)d_in[3];
    const float* priors      = (const float*)d_in[4];
    float* out = (float*)d_out;

    const size_t BP = (size_t)BB * PP;
    char* ws = (char*)d_ws;
    float*  ws_ovl = (float*)(ws);                         // B*P f32 -> conf_all
    int*    ws_obj = (int*)(ws + BP * 4);                  // B*P i32 -> gt_label
    float4* ws_loc = (float4*)(ws + BP * 8);               // B*P*4 f32
    float*  ws_bt  = (float*)(ws + BP * 8 + BP * 16);      // B f32

    ssd_match_kernel<<<BB, TPB, 0, stream>>>(boxes, labels, priors,
                                             ws_ovl, ws_obj, ws_loc);
    ssd_conf_kernel<<<NTILES / WAVES_PER_BLK, TPB, 0, stream>>>(pred_scores,
                                                                ws_obj, ws_ovl);
    ssd_loss_kernel<<<BB, TPB, 0, stream>>>(pred_locs, ws_ovl, ws_obj,
                                            ws_loc, ws_bt);
    ssd_mean_kernel<<<1, BB, 0, stream>>>(ws_bt, out);
}
